// Int4GemmPluginModule_82738249990454
// MI455X (gfx1250) — matrix operand order repaired and checked by reading, not compiled, not run on hardware
//
#include <hip/hip_runtime.h>

// Problem constants (fixed by the reference).
constexpr int Kdim = 4096;
constexpr int Ndim = 11008;
constexpr int Mdim = 8192;      // B*S = 4*2048
constexpr int Gsz  = 128;       // quant group size

typedef __attribute__((ext_vector_type(16))) __bf16 bf16x16;
typedef __attribute__((ext_vector_type(8)))  float  floatx8;

union AFrag { bf16x16 v; uint4 q[2]; unsigned int u[8]; };
union BFrag { bf16x16 v; uint4 q[2]; unsigned int u[8]; };

__device__ __forceinline__ unsigned int pk_bf16_hw(float lo, float hi) {
    unsigned short l = __builtin_bit_cast(unsigned short, (__bf16)lo);
    unsigned short h = __builtin_bit_cast(unsigned short, (__bf16)hi);
    return (unsigned int)l | ((unsigned int)h << 16);
}

// CDNA5 async global->LDS copy (ASYNCcnt), 16B per lane, per-lane addresses.
__device__ __forceinline__ void async_copy_b128(unsigned lds_addr,
                                                unsigned long long gaddr) {
    asm volatile("global_load_async_to_lds_b128 %0, %1, off"
                 :: "v"(lds_addr), "v"(gaddr) : "memory");
}
__device__ __forceinline__ void wait_async0() {
    asm volatile("s_wait_asynccnt 0x0" ::: "memory");
}

// ============================================================================
// Fast path: prepasses + clean bf16 WMMA GEMM (needs ~150 MiB workspace).
// ws layout: [0, M*K*2)        : x as bf16, row-major (M, K)
//            [M*K*2, +N*K/2*4) : weights bf16, TRANSPOSED (N, K/2) K-pair
//                                dwords: dword (n, kk) = {bf16 k=2kk, 2kk+1}
// ============================================================================
constexpr size_t WSX_BYTES = (size_t)Mdim * Kdim * 2;          // 64 MiB
constexpr size_t WSW_BYTES = (size_t)Ndim * (Kdim / 2) * 4;    // ~86 MiB

__global__ __launch_bounds__(256)
void xcvt_bf16(const float* __restrict__ x, unsigned short* __restrict__ xb) {
    const size_t base = ((size_t)blockIdx.x * 256 + threadIdx.x) * 8;
    const float4 a = *(const float4*)(x + base);
    const float4 b = *(const float4*)(x + base + 4);
    uint4 d;
    d.x = pk_bf16_hw(a.x, a.y);
    d.y = pk_bf16_hw(a.z, a.w);
    d.z = pk_bf16_hw(b.x, b.y);
    d.w = pk_bf16_hw(b.z, b.w);
    *(uint4*)(xb + base) = d;
}

__global__ __launch_bounds__(256)
void wdeq_bf16_t(const int* __restrict__ qw, const float* __restrict__ sc,
                 unsigned int* __restrict__ wp) {
    const int n   = blockIdx.x * 256 + threadIdx.x;   // 0..11007
    const int kr4 = blockIdx.y;                        // 4 packed rows: 0..127
    const float s  = sc[(size_t)(kr4 >> 2) * Ndim + n]; // group = 32*kr4/128
    const float s8 = 8.0f * s;
    unsigned int d[16];
#pragma unroll
    for (int j = 0; j < 4; ++j) {
        const int q = qw[(size_t)(4 * kr4 + j) * Ndim + n];
#pragma unroll
        for (int p = 0; p < 4; ++p) {
            const float f0 = (float)((q >> (8 * p))     & 15) * s - s8;
            const float f1 = (float)((q >> (8 * p + 4)) & 15) * s - s8;
            d[4 * j + p] = pk_bf16_hw(f0, f1);
        }
    }
    // 64B contiguous per thread at transposed location (n, 16*kr4 ...)
    unsigned int* dst = wp + (size_t)n * (Kdim / 2) + 16 * kr4;
#pragma unroll
    for (int i = 0; i < 4; ++i) *(uint4*)(dst + 4 * i) = *(uint4*)&d[4 * i];
}

// ---- main GEMM tiling ----
constexpr int BM = 128;
constexpr int BN = 128;
constexpr int BK = 64;                  // two bf16 WMMA K-steps per chunk
constexpr int NCHUNK = Kdim / BK;       // 64
constexpr int AS_STR = BK + 8;          // 72 shorts (144B rows, 16B-aligned)
constexpr int BS_STR = BK / 2 + 4;      // 36 dwords (144B rows, 16B-aligned)

__global__ __launch_bounds__(256)
void bf16gemm_wmma(const unsigned short* __restrict__ xb,
                   const unsigned int*  __restrict__ wp,
                   float* __restrict__ out) {
    __shared__ __align__(16) unsigned short As[2][BM][AS_STR];  // 36.0 KB
    __shared__ __align__(16) unsigned int   Bs[2][BN][BS_STR];  // 36.0 KB

    const int tid  = threadIdx.x;
    const int lane = tid & 31;
    const int wave = tid >> 5;
    const int wm   = wave >> 2;          // 0..1 : 64-row M slice
    const int wn   = wave & 3;           // 0..3 : 32-col N slice
    const int half = lane >> 4;
    const int l16  = lane & 15;

    const int m0 = blockIdx.y * BM;
    const int n0 = blockIdx.x * BN;

    // async DMA of one chunk into LDS buffer `buf` (no VGPR staging)
    auto stage_async = [&](int kb, int buf) {
#pragma unroll
        for (int i = 0; i < 4; ++i) {                // A: 1024 x 16B / 256 thr
            const int idx = tid + 256 * i;
            const int r   = idx >> 3;                // 0..127
            const int c   = idx & 7;                 // 8 bf16 per 16B chunk
            async_copy_b128(
                (unsigned)(size_t)&As[buf][r][c * 8],
                (unsigned long long)(xb + (size_t)(m0 + r) * Kdim + kb + c * 8));
        }
        const int kk0 = kb >> 1;
#pragma unroll
        for (int i = 0; i < 4; ++i) {                // B: 1024 x 16B / 256 thr
            const int idx = tid + 256 * i;
            const int r   = idx >> 3;                // n row 0..127
            const int c   = idx & 7;                 // 4 dwords per 16B chunk
            async_copy_b128(
                (unsigned)(size_t)&Bs[buf][r][c * 4],
                (unsigned long long)(wp + (size_t)(n0 + r) * (Kdim / 2) + kk0 + c * 4));
        }
    };

    floatx8 acc[4][2] = {};                          // 4 M-tiles x 2 N-tiles

    stage_async(0, 0);
    wait_async0();
    __syncthreads();

    int buf = 0;
    for (int c = 0; c < NCHUNK; ++c) {
        if (c + 1 < NCHUNK) stage_async((c + 1) * BK, buf ^ 1);

#pragma unroll
        for (int s = 0; s < 2; ++s) {                // two WMMA K-steps
            AFrag a[4];
#pragma unroll
            for (int t = 0; t < 4; ++t) {
                const unsigned short* arow = &As[buf][wm * 64 + t * 16 + l16][0];
                a[t].q[0] = *(const uint4*)(arow + s * 32 + 8 * half);
                a[t].q[1] = *(const uint4*)(arow + s * 32 + 16 + 8 * half);
            }
            BFrag b[2];
#pragma unroll
            for (int u = 0; u < 2; ++u) {
                const unsigned int* brow = &Bs[buf][wn * 32 + u * 16 + l16][0];
                b[u].q[0] = *(const uint4*)(brow + s * 16 + 4 * half);
                b[u].q[1] = *(const uint4*)(brow + s * 16 + 8 + 4 * half);
            }
#pragma unroll
            for (int t = 0; t < 4; ++t)
#pragma unroll
                for (int u = 0; u < 2; ++u)
                    acc[t][u] = __builtin_amdgcn_wmma_f32_16x16x32_bf16(
                        false, a[t].v, false, b[u].v, (short)0, acc[t][u],
                        false, false);
        }

        if (c + 1 < NCHUNK) wait_async0();
        __syncthreads();
        buf ^= 1;
    }

#pragma unroll
    for (int t = 0; t < 4; ++t)
#pragma unroll
        for (int u = 0; u < 2; ++u)
#pragma unroll
            for (int r = 0; r < 8; ++r) {
                const int m = m0 + wm * 64 + t * 16 + half * 8 + r;
                const int n = n0 + wn * 32 + u * 16 + l16;
                out[(size_t)m * Ndim + n] = acc[t][u][r];
            }
}

// ============================================================================
// Fallback: fused dequant+GEMM (round-1 kernel), used when ws is too small.
// ============================================================================
constexpr int FBK     = 32;
constexpr int FNCHUNK = Kdim / FBK;
constexpr int FA_STR  = FBK + 8;
constexpr int FB_STR  = 2 * BN + 8;

__device__ __forceinline__ unsigned int pack2_bf16(float lo, float hi) {
    unsigned int ulo = __float_as_uint(lo);
    unsigned int uhi = __float_as_uint(hi);
    ulo = (ulo + 0x7FFFu + ((ulo >> 16) & 1u)) >> 16;
    uhi = (uhi + 0x7FFFu + ((uhi >> 16) & 1u)) >> 16;
    return (ulo & 0xFFFFu) | (uhi << 16);
}

__global__ __launch_bounds__(256)
void int4gemm_fused(const float* __restrict__ x, const int* __restrict__ qw,
                    const float* __restrict__ sc, float* __restrict__ out) {
    __shared__ unsigned short As[2][BM][FA_STR];
    __shared__ unsigned short Bs[2][FBK / 2][FB_STR];

    const int tid  = threadIdx.x;
    const int lane = tid & 31;
    const int wave = tid >> 5;
    const int wm = wave >> 2, wn = wave & 3;
    const int half = lane >> 4, l16 = lane & 15;
    const int m0 = blockIdx.y * BM;
    const int n0 = blockIdx.x * BN;

    float4 xr[4];
    int    qr[2];
    float  srg[2];

    auto global_stage = [&](int kb) {
#pragma unroll
        for (int i = 0; i < 4; ++i) {
            const int idx = tid + 256 * i;
            xr[i] = *(const float4*)(x + (size_t)(m0 + (idx >> 3)) * Kdim + kb +
                                     (idx & 7) * 4);
        }
        const int kr0 = kb >> 3, g = kb / Gsz;
#pragma unroll
        for (int i = 0; i < 2; ++i) {
            const int idx = tid + 256 * i;
            qr[i]  = qw[(size_t)(kr0 + (idx >> 7)) * Ndim + n0 + (idx & 127)];
            srg[i] = sc[(size_t)g * Ndim + n0 + (idx & 127)];
        }
    };
    auto lds_commit = [&](int buf) {
#pragma unroll
        for (int i = 0; i < 4; ++i) {
            const int idx = tid + 256 * i;
            unsigned short* dst = &As[buf][idx >> 3][(idx & 7) * 4];
            *(unsigned int*)(dst)     = pack2_bf16(xr[i].x, xr[i].y);
            *(unsigned int*)(dst + 2) = pack2_bf16(xr[i].z, xr[i].w);
        }
#pragma unroll
        for (int i = 0; i < 2; ++i) {
            const int idx = tid + 256 * i;
            const int qrow = idx >> 7, n = idx & 127;
            const int q = qr[i];
            const float s = srg[i];
#pragma unroll
            for (int p = 0; p < 4; ++p) {
                const float f0 = (float)(((q >> (8 * p))     & 15) - 8) * s;
                const float f1 = (float)(((q >> (8 * p + 4)) & 15) - 8) * s;
                *(unsigned int*)&Bs[buf][4 * qrow + p][2 * n] = pack2_bf16(f0, f1);
            }
        }
    };

    floatx8 acc[4][2] = {};
    global_stage(0);
    lds_commit(0);
    int buf = 0;
    for (int c = 0; c < FNCHUNK; ++c) {
        __syncthreads();
        if (c + 1 < FNCHUNK) global_stage((c + 1) * FBK);
        AFrag a[4];
#pragma unroll
        for (int t = 0; t < 4; ++t) {
            const unsigned short* arow = &As[buf][wm * 64 + t * 16 + l16][0];
#pragma unroll
            for (int v = 0; v < 8; ++v) {
                const int kbase = ((v < 4) ? 2 * v : 2 * v + 8) + 8 * half;
                a[t].u[v] = *(const unsigned int*)(arow + kbase);
            }
        }
        BFrag b[2];
#pragma unroll
        for (int u = 0; u < 2; ++u) {
            const int n = wn * 32 + u * 16 + l16;
#pragma unroll
            for (int v = 0; v < 8; ++v)
                b[u].u[v] = *(const unsigned int*)&Bs[buf][((v < 4) ? v : v + 4) + 4 * half][2 * n];
        }
#pragma unroll
        for (int t = 0; t < 4; ++t)
#pragma unroll
            for (int u = 0; u < 2; ++u)
                acc[t][u] = __builtin_amdgcn_wmma_f32_16x16x32_bf16(
                    false, a[t].v, false, b[u].v, (short)0, acc[t][u], false, false);
        if (c + 1 < FNCHUNK) lds_commit(buf ^ 1);
        buf ^= 1;
    }
#pragma unroll
    for (int t = 0; t < 4; ++t)
#pragma unroll
        for (int u = 0; u < 2; ++u)
#pragma unroll
            for (int r = 0; r < 8; ++r) {
                const int m = m0 + wm * 64 + t * 16 + half * 8 + r;
                const int n = n0 + wn * 32 + u * 16 + l16;
                out[(size_t)m * Ndim + n] = acc[t][u][r];
            }
}

// ============================================================================

extern "C" void kernel_launch(void* const* d_in, const int* in_sizes, int n_in,
                              void* d_out, int out_size, void* d_ws, size_t ws_size,
                              hipStream_t stream) {
    (void)in_sizes; (void)n_in; (void)out_size;
    const float* x   = (const float*)d_in[0];
    const int*   qw  = (const int*)d_in[1];
    const float* sc  = (const float*)d_in[2];
    float*       out = (float*)d_out;

    if (ws_size >= WSX_BYTES + WSW_BYTES) {
        unsigned short* wsx = (unsigned short*)d_ws;
        unsigned int*   wsw = (unsigned int*)((char*)d_ws + WSX_BYTES);

        xcvt_bf16<<<(size_t)Mdim * Kdim / (256 * 8), 256, 0, stream>>>(x, wsx);
        wdeq_bf16_t<<<dim3(Ndim / 256, Kdim / 32), 256, 0, stream>>>(qw, sc, wsw);
        bf16gemm_wmma<<<dim3(Ndim / BN, Mdim / BM), 256, 0, stream>>>(wsx, wsw, out);
    } else {
        int4gemm_fused<<<dim3(Ndim / BN, Mdim / BM), 256, 0, stream>>>(x, qw, sc, out);
    }
}
